// ModelPaperBaseline_bin_39359080300742
// MI455X (gfx1250) — 8-bit matrix operand order repaired and checked by reading, compiled, not run on hardware
//
#include <hip/hip_runtime.h>
#include <cstdint>

typedef __attribute__((ext_vector_type(2)))  float    v2f;
typedef __attribute__((ext_vector_type(8)))  float    v8f;
typedef __attribute__((ext_vector_type(8)))  int      v8i;
typedef __attribute__((ext_vector_type(4)))  int      iv4;
typedef __attribute__((ext_vector_type(2)))  int      iv2;
typedef __attribute__((ext_vector_type(4)))  unsigned u32x4;
typedef __attribute__((ext_vector_type(8)))  int      i32x8;
typedef __attribute__((ext_vector_type(4)))  int      i32x4;

#define HAVE_WMMA_F32X4 __has_builtin(__builtin_amdgcn_wmma_f32_16x16x4_f32)
#define HAVE_TDM        __has_builtin(__builtin_amdgcn_tensor_load_to_lds)

static constexpr int L_WC    = 0;
static constexpr int L_BFC1  = 16384;
static constexpr int L_BFC2  = 49152;
static constexpr int L_BN0   = 53248;
static constexpr int L_BNS   = 53504;
static constexpr int L_BN5   = 54528;
static constexpr int L_BN6   = 55040;
static constexpr int L_WAVE  = 55552;
static constexpr int W_XBUF  = 0;
static constexpr int W_YPAD  = 4096;
static constexpr int W_SC    = 14336;
static constexpr int W_YT    = 22528;
static constexpr int W_H1    = 30720;
static constexpr int W_H2    = 31744;
static constexpr int W_STRIDE= 32768;
static constexpr int LDS_TOTAL = L_WAVE + 8 * W_STRIDE;

struct Params {
  const float *x, *conv0_w, *conv0_b, *bn0_g, *bn0_b, *bn0_m, *bn0_v;
  const float *convs_w, *convs_b, *bns_g, *bns_b, *bns_m, *bns_v;
  const float *fc1_w, *fc1_b, *bn5_g, *bn5_b, *bn5_m, *bn5_v;
  const float *fc2_w, *fc2_b, *bn6_g, *bn6_b, *bn6_m, *bn6_v;
  const float *fc3_w, *fc3_b;
  float *out;
};

__device__ __forceinline__ signed char sgn8(float w) { return w >= 0.f ? (signed char)1 : (signed char)-1; }
__device__ __forceinline__ float       sgnf(float w) { return w >= 0.f ? 1.f : -1.f; }
__device__ __forceinline__ v8i         v8i_zero()    { v8i z = {0,0,0,0,0,0,0,0}; return z; }

__device__ __forceinline__ v8i frag_a_row(const char* row, int h) {
  iv2 d0 = *(const iv2*)(row +  0 + 16*h);
  iv2 d1 = *(const iv2*)(row +  8 + 16*h);
  iv2 d2 = *(const iv2*)(row + 32 + 16*h);
  iv2 d3 = *(const iv2*)(row + 40 + 16*h);
  v8i f = { d0[0], d0[1], d1[0], d1[1], d2[0], d2[1], d3[0], d3[1] };
  return f;
}

__device__ __forceinline__ v8i frag_b(iv4 lo, iv4 hi) {
  v8i f = { lo[0], lo[1], lo[2], lo[3], hi[0], hi[1], hi[2], hi[3] };
  return f;
}

__global__ void __launch_bounds__(256, 1)
bnn_fused_gfx1250(Params p, int Btot) {
  extern __shared__ char sm[];
  const int tid = threadIdx.x;

  for (int i = tid; i < 4*32*128; i += 256) {
    int l = i >> 12, rem = i & 4095, co = rem >> 7, K = rem & 127;
    int k = K >> 5, cin = K & 31;
    signed char v = 0;
    if (k < 3) v = sgn8(p.convs_w[((l*32 + co)*32 + cin)*3 + k]);
    sm[L_WC + i] = v;
  }
  for (int i = tid; i < 32768; i += 256) {
    int b = i & 3, v = (i >> 2) & 7, ln = (i >> 5) & 31, ch = (i >> 10) & 7, nt = i >> 13;
    int nn = nt*16 + (ln & 15), hh = ln >> 4;
    int K = 64*ch + 32*(v >> 2) + 16*hh + 4*(v & 3) + b;
    sm[L_BFC1 + i] = sgn8(p.fc1_w[nn*512 + K]);
  }
  for (int i = tid; i < 4096; i += 256) {
    int b = i & 3, v = (i >> 2) & 7, ln = (i >> 5) & 31, nt = i >> 10;
    int nn = nt*16 + (ln & 15), hh = ln >> 4;
    int K = 32*(v >> 2) + 16*hh + 4*(v & 3) + b;
    sm[L_BFC2 + i] = sgn8(p.fc2_w[nn*64 + K]);
  }
  if (tid < 32) {
    float s = p.bn0_g[tid] * rsqrtf(p.bn0_v[tid] + 0.01f);
    *(float2*)(sm + L_BN0 + tid*8) = make_float2(s, (p.conv0_b[tid] - p.bn0_m[tid])*s + p.bn0_b[tid]);
  }
  if (tid < 128) {
    float s = p.bns_g[tid] * rsqrtf(p.bns_v[tid] + 0.01f);
    *(float2*)(sm + L_BNS + tid*8) = make_float2(s, (p.convs_b[tid] - p.bns_m[tid])*s + p.bns_b[tid]);
  }
  if (tid < 64) {
    float s = p.bn5_g[tid] * rsqrtf(p.bn5_v[tid] + 0.01f);
    *(float2*)(sm + L_BN5 + tid*8) = make_float2(s, (p.fc1_b[tid] - p.bn5_m[tid])*s + p.bn5_b[tid]);
    float s6 = p.bn6_g[tid] * rsqrtf(p.bn6_v[tid] + 0.01f);
    *(float2*)(sm + L_BN6 + tid*8) = make_float2(s6, (p.fc2_b[tid] - p.bn6_m[tid])*s6 + p.bn6_b[tid]);
  }
  __syncthreads();

  const int wave = tid >> 5, lane = tid & 31;
  const int n = lane & 15, h = lane >> 4;
  const int sBase = (int)blockIdx.x * 128 + wave * 16;
  if (sBase + 16 > Btot) return;
  char* wv = sm + L_WAVE + wave * W_STRIDE;

#if HAVE_TDM
  {
    uint64_t ga = (uint64_t)(uintptr_t)(p.x + (size_t)sBase * 64);
    unsigned la = (unsigned)(uintptr_t)(wv + W_XBUF);
    u32x4 g0 = { 1u, la, (unsigned)ga,
                 (unsigned)(((ga >> 32) & 0x01FFFFFFull) | 0x80000000u) };
    i32x8 g1 = { 0x00020000, (int)(1024u << 16), 0x00010000, (int)(1024u << 16),
                 1, 1024, 0, 0 };
    i32x4 g2 = {0, 0, 0, 0};
    i32x4 g3 = {0, 0, 0, 0};
    i32x8 g4 = {0, 0, 0, 0, 0, 0, 0, 0};
    __builtin_amdgcn_tensor_load_to_lds(g0, g1, g2, g3, g4, 0);
    __builtin_amdgcn_s_wait_tensorcnt(0);
  }
#else
  {
    const float* xg = p.x + (size_t)sBase * 64;
    float* xl = (float*)(wv + W_XBUF);
    for (int i = lane; i < 1024; i += 32) xl[i] = xg[i];
  }
#endif

  {
    int zs = lane & 15, zr = (lane >> 4) ? 17 : 0;
    iv4 z = {0, 0, 0, 0};
    *(iv4*)(wv + W_YPAD + zs*640 + zr*32)      = z;
    *(iv4*)(wv + W_YPAD + zs*640 + zr*32 + 16) = z;
  }

  {
    float scl[2][8], shf[2][8];
#pragma unroll
    for (int t = 0; t < 2; t++) {
      const float2* bp = (const float2*)(sm + L_BN0 + (t*16 + 8*h)*8);
#pragma unroll
      for (int r = 0; r < 8; r++) { scl[t][r] = bp[r].x; shf[t][r] = bp[r].y; }
    }
#if HAVE_WMMA_F32X4
    v2f a0[2];
#pragma unroll
    for (int t = 0; t < 2; t++) {
      a0[t][0] = sgnf(p.conv0_w[(t*16 + n)*4 + (0 + 2*h)]);
      a0[t][1] = sgnf(p.conv0_w[(t*16 + n)*4 + (1 + 2*h)]);
    }
#endif
#pragma unroll 1
    for (int s = 0; s < 16; s++) {
      const float* xs = (const float*)(wv + W_XBUF) + s*64;
      v8f acc[2];
#if HAVE_WMMA_F32X4
      v2f bx; bx[0] = xs[(0 + 2*h)*16 + n]; bx[1] = xs[(1 + 2*h)*16 + n];
#pragma unroll
      for (int t = 0; t < 2; t++) {
        v8f c0 = {0.f,0.f,0.f,0.f,0.f,0.f,0.f,0.f};
        acc[t] = __builtin_amdgcn_wmma_f32_16x16x4_f32(false, a0[t], false, bx,
                                                       (short)0, c0, false, false);
      }
#else
#pragma unroll
      for (int t = 0; t < 2; t++) {
#pragma unroll
        for (int r = 0; r < 8; r++) {
          int c = t*16 + 8*h + r;
          float sum = 0.f;
#pragma unroll
          for (int ci = 0; ci < 4; ci++) sum += sgnf(p.conv0_w[c*4 + ci]) * xs[ci*16 + n];
          acc[t][r] = sum;
        }
      }
#endif
#pragma unroll
      for (int t = 0; t < 2; t++) {
        unsigned lo = 0, hi = 0;
#pragma unroll
        for (int r = 0; r < 8; r++) {
          float f = acc[t][r]*scl[t][r] + shf[t][r];
          unsigned by = (unsigned char)sgn8(f);
          if (r < 4) lo |= by << (8*r); else hi |= by << (8*(r - 4));
        }
        iv2 pk; pk[0] = (int)lo; pk[1] = (int)hi;
        *(iv2*)(wv + W_YPAD + s*640 + (n + 1)*32 + t*16 + 8*h) = pk;
        *(iv2*)(wv + W_SC   + s*512 +  n*32      + t*16 + 8*h) = pk;
      }
    }
  }

#pragma unroll 1
  for (int l = 0; l < 4; l++) {
    v8i aW[2][2];
    float scl[2][8], shf[2][8];
#pragma unroll
    for (int t = 0; t < 2; t++) {
      const char* wrow = sm + L_WC + (l*32 + t*16 + n)*128;
      aW[t][0] = frag_a_row(wrow,      h);
      aW[t][1] = frag_a_row(wrow + 64, h);
      const float2* bp = (const float2*)(sm + L_BNS + (l*32 + t*16 + 8*h)*8);
#pragma unroll
      for (int r = 0; r < 8; r++) { scl[t][r] = bp[r].x; shf[t][r] = bp[r].y; }
    }
#pragma unroll 1
    for (int s = 0; s < 16; s++) {
      const char* yp = wv + W_YPAD + s*640;
      iv4 r0 = *(const iv4*)(yp + (n + 0)*32 + 16*h);
      iv4 r1 = *(const iv4*)(yp + (n + 1)*32 + 16*h);
      iv4 r2 = *(const iv4*)(yp + (n + 2)*32 + 16*h);
      iv4 zz = {0, 0, 0, 0};
      v8i b0 = frag_b(r0, r1);
      v8i b1 = frag_b(r2, zz);
      v8i acc[2];
#pragma unroll
      for (int t = 0; t < 2; t++) {
        v8i c0 = v8i_zero();
        c0     = __builtin_amdgcn_wmma_i32_16x16x64_iu8(true, aW[t][0], true, b0, c0, false, false);
        acc[t] = __builtin_amdgcn_wmma_i32_16x16x64_iu8(true, aW[t][1], true, b1, c0, false, false);
      }
#pragma unroll
      for (int t = 0; t < 2; t++) {
        iv2 scv = *(const iv2*)(wv + W_SC + s*512 + n*32 + t*16 + 8*h);
        unsigned lo = 0, hi = 0;
#pragma unroll
        for (int r = 0; r < 8; r++) {
          float f = (float)acc[t][r]*scl[t][r] + shf[t][r];
          int sb = (f >= 0.f) ? 1 : -1;
          int sc = (int)(signed char)((((unsigned)((r < 4) ? scv[0] : scv[1])) >> (8*(r & 3))) & 0xFFu);
          int y  = sb + sc;
          if (l < 3) {
            unsigned by = (unsigned char)(signed char)y;
            if (r < 4) lo |= by << (8*r); else hi |= by << (8*(r - 4));
          } else {
            *(wv + W_YT + s*512 + (t*16 + 8*h + r)*16 + n) = (char)y;
          }
        }
        if (l < 3) {
          iv2 pk; pk[0] = (int)lo; pk[1] = (int)hi;
          *(iv2*)(wv + W_YPAD + s*640 + (n + 1)*32 + t*16 + 8*h) = pk;
        }
      }
    }
  }

  v8i afc[8];
#pragma unroll
  for (int ch = 0; ch < 8; ch++)
    afc[ch] = frag_a_row(wv + W_YT + n*512 + ch*64, h);
#pragma unroll
  for (int nt = 0; nt < 4; nt++) {
    v8i acc = v8i_zero();
#pragma unroll
    for (int ch = 0; ch < 8; ch++) {
      const char* bpt = sm + L_BFC1 + ((nt*8 + ch)*32 + lane)*32;
      v8i bf = frag_b(*(const iv4*)bpt, *(const iv4*)(bpt + 16));
      acc = __builtin_amdgcn_wmma_i32_16x16x64_iu8(true, afc[ch], true, bf, acc, false, false);
    }
    int u = nt*16 + n;
    float2 bn = *(const float2*)(sm + L_BN5 + u*8);
#pragma unroll
    for (int r = 0; r < 8; r++) {
      float f = (float)acc[r]*bn.x + bn.y;
      *(wv + W_H1 + (r + 8*h)*64 + u) = sgn8(f);
    }
  }

  v8i a2 = frag_a_row(wv + W_H1 + n*64, h);
#pragma unroll
  for (int nt = 0; nt < 4; nt++) {
    const char* bpt = sm + L_BFC2 + (nt*32 + lane)*32;
    v8i bf = frag_b(*(const iv4*)bpt, *(const iv4*)(bpt + 16));
    v8i acc = __builtin_amdgcn_wmma_i32_16x16x64_iu8(true, a2, true, bf, v8i_zero(), false, false);
    int u = nt*16 + n;
    float2 bn = *(const float2*)(sm + L_BN6 + u*8);
#pragma unroll
    for (int r = 0; r < 8; r++) {
      float f = (float)acc[r]*bn.x + bn.y;
      *(wv + W_H2 + (r + 8*h)*64 + u) = sgn8(f);
    }
  }

  {
    const char* hrow = wv + W_H2 + n*64 + 32*h;
    iv4 q0 = *(const iv4*)hrow;
    iv4 q1 = *(const iv4*)(hrow + 16);
    float part = 0.f;
#pragma unroll
    for (int j = 0; j < 8; j++) {
      int d  = (j < 4) ? q0[j] : q1[j - 4];
      int ub = 32*h + j*4;
#pragma unroll
      for (int bb = 0; bb < 4; bb++) {
        int val = (int)(signed char)((d >> (8*bb)) & 0xFF);
        part += sgnf(p.fc3_w[ub + bb]) * (float)val;
      }
    }
    float* red = (float*)(wv + W_H1);
    red[lane] = part;
    asm volatile("" ::: "memory");
    float tot = red[n] + red[n + 16];
    if (lane < 16) {
      float t = tot + p.fc3_b[0];
      p.out[sBase + lane] = 1.f / (1.f + __expf(-t));
    }
  }
}

extern "C" void kernel_launch(void* const* d_in, const int* in_sizes, int n_in,
                              void* d_out, int out_size, void* d_ws, size_t ws_size,
                              hipStream_t stream) {
  (void)n_in; (void)out_size; (void)d_ws; (void)ws_size;
  Params p;
  p.x       = (const float*)d_in[0];
  p.conv0_w = (const float*)d_in[1];  p.conv0_b = (const float*)d_in[2];
  p.bn0_g   = (const float*)d_in[3];  p.bn0_b   = (const float*)d_in[4];
  p.bn0_m   = (const float*)d_in[5];  p.bn0_v   = (const float*)d_in[6];
  p.convs_w = (const float*)d_in[7];  p.convs_b = (const float*)d_in[8];
  p.bns_g   = (const float*)d_in[9];  p.bns_b   = (const float*)d_in[10];
  p.bns_m   = (const float*)d_in[11]; p.bns_v   = (const float*)d_in[12];
  p.fc1_w   = (const float*)d_in[13]; p.fc1_b   = (const float*)d_in[14];
  p.bn5_g   = (const float*)d_in[15]; p.bn5_b   = (const float*)d_in[16];
  p.bn5_m   = (const float*)d_in[17]; p.bn5_v   = (const float*)d_in[18];
  p.fc2_w   = (const float*)d_in[19]; p.fc2_b   = (const float*)d_in[20];
  p.bn6_g   = (const float*)d_in[21]; p.bn6_b   = (const float*)d_in[22];
  p.bn6_m   = (const float*)d_in[23]; p.bn6_v   = (const float*)d_in[24];
  p.fc3_w   = (const float*)d_in[25]; p.fc3_b   = (const float*)d_in[26];
  p.out     = (float*)d_out;

  int Btot   = in_sizes[0] / 64;
  int blocks = (Btot + 127) / 128;
  bnn_fused_gfx1250<<<blocks, 256, LDS_TOTAL, stream>>>(p, Btot);
}
